// LearnedRouterModel_34583076667907
// MI455X (gfx1250) — matrix operand order repaired and checked by reading, not compiled, not run on hardware
//
#include <hip/hip_runtime.h>
#include <hip/hip_bf16.h>

typedef __attribute__((ext_vector_type(16))) _Float16 v16h;
typedef __attribute__((ext_vector_type(8)))  float    v8f;

#define Hh   64
#define Vv   64
#define HALF 32
#define Bn   256
#define Ln   2048

// ---------------- WMMA fragment loaders (CDNA5 layouts, wave32) ----------------
// 16-bit A 16x32 (MxK): lanes 0-15 rows M, K{0..7}in v0-3,{16..23}in v4-7; lanes16-31: K+8
__device__ __forceinline__ v16h load_a_frag(const _Float16* S, int ld, int r0, int k0, int lane){
  v16h a;
  int m = lane & 15;
  int kbase = (lane >> 4) << 3;          // 0 or 8
#pragma unroll
  for (int h = 0; h < 16; ++h){
    int vv = h >> 1, hh = h & 1;
    int k = ((vv & 3) * 2 + hh) + ((vv >= 4) ? 16 : 0) + kbase;
    a[h] = S[(r0 + m) * ld + k0 + k];
  }
  return a;
}
// 16-bit B 32x16 (KxN): lane = N; lanes 0-15 hold K{0..7},{16..23}; lanes16-31: K+8
__device__ __forceinline__ v16h load_b_frag(const _Float16* S, int ld, int k0, int c0, int lane){
  v16h b;
  int n = lane & 15;
  int kbase = (lane >> 4) << 3;          // 0 or 8
#pragma unroll
  for (int h = 0; h < 16; ++h){
    b[h] = S[(k0 + kbase + ((h < 8) ? h : h + 8)) * ld + c0 + n];
  }
  return b;
}

// ---------------- Kernel 1: build 64-entry token table (WMMA) ----------------
// ws_ks[64][32] = token@sem_w+sem_b ; ws_ke likewise ; ws_scal[64][4] = {g, 1-g, 1/(ss+1e-6), 1/(se+1e-6)}
__global__ __launch_bounds__(256) void build_table_kernel(
    const float* __restrict__ embed, const float* __restrict__ ff_w1, const float* __restrict__ ff_b1,
    const float* __restrict__ ff_w2, const float* __restrict__ ff_b2,
    const float* __restrict__ ln_g,  const float* __restrict__ ln_b,
    const float* __restrict__ sem_w, const float* __restrict__ sem_b,
    const float* __restrict__ epi_w, const float* __restrict__ epi_b,
    const float* __restrict__ rtr_w, const float* __restrict__ rtr_b,
    float* __restrict__ ws_ks, float* __restrict__ ws_ke, float* __restrict__ ws_scal)
{
  __shared__ __align__(16) unsigned char smem1[57600];
  _Float16* EH = (_Float16*)(smem1);           // [0,8K)   embed f16, 64x64 ld=64
  _Float16* WB = (_Float16*)(smem1 + 8192);    // [8K,24K) W1(64x128)/W2(128x64)/WP(64x80) f16
  _Float16* RC = (_Float16*)(smem1 + 24576);   // [24K,40K) relu(64x128) then h(64x64) f16
  float*    XD = (float*)(smem1 + 40960);      // [40K,56K) x f32 64x64 then Pks(2048)+Pke(2048)
  float*    GB = (float*)(smem1 + 57344);      // [56K,56.25K) g per token

  const int tid  = threadIdx.x;
  const int lane = tid & 31;
  const int wv   = tid >> 5;

  for (int i = tid; i < 64*64;  i += 256) EH[i] = (_Float16)embed[i];
  for (int i = tid; i < 64*128; i += 256) WB[i] = (_Float16)ff_w1[i];
  __syncthreads();

  // ---- matmul1: e(64x64) @ w1(64x128) + b1, relu -> RC (ld=128)
#pragma unroll
  for (int tt = 0; tt < 4; ++tt){
    int tile = wv + tt*8;                       // 0..31 (4x8 tiles)
    int r0 = (tile >> 3) * 16, c0 = (tile & 7) * 16;
    v8f c = {};
#pragma unroll
    for (int ks = 0; ks < 64; ks += 32){
      v16h a = load_a_frag(EH, 64, r0, ks, lane);
      v16h b = load_b_frag(WB, 128, ks, c0, lane);
      c = __builtin_amdgcn_wmma_f32_16x16x32_f16(false, a, false, b, (short)0, c, false, false);
    }
    int n = lane & 15, mh = (lane >> 4) << 3;
#pragma unroll
    for (int r = 0; r < 8; ++r){
      int m = r0 + r + mh, col = c0 + n;
      float vsum = c[r] + ff_b1[col];
      RC[m * 128 + col] = (_Float16)(vsum > 0.f ? vsum : 0.f);
    }
  }
  __syncthreads();

  for (int i = tid; i < 128*64; i += 256) WB[i] = (_Float16)ff_w2[i];
  __syncthreads();

  // ---- matmul2: relu(64x128) @ w2(128x64) + b2 + e -> XD f32 (ld=64)
#pragma unroll
  for (int tt = 0; tt < 2; ++tt){
    int tile = wv + tt*8;                       // 0..15 (4x4 tiles)
    int r0 = (tile >> 2) * 16, c0 = (tile & 3) * 16;
    v8f c = {};
#pragma unroll
    for (int ks = 0; ks < 128; ks += 32){
      v16h a = load_a_frag(RC, 128, r0, ks, lane);
      v16h b = load_b_frag(WB, 64, ks, c0, lane);
      c = __builtin_amdgcn_wmma_f32_16x16x32_f16(false, a, false, b, (short)0, c, false, false);
    }
    int n = lane & 15, mh = (lane >> 4) << 3;
#pragma unroll
    for (int r = 0; r < 8; ++r){
      int m = r0 + r + mh, col = c0 + n;
      XD[m * 64 + col] = c[r] + ff_b2[col] + (float)EH[m * 64 + col];
    }
  }
  __syncthreads();

  // ---- LayerNorm rows (threads 0..63) -> h f16 into RC; stage WP(64x80) into WB
  if (tid < 64){
    float s = 0.f, s2 = 0.f;
#pragma unroll
    for (int j = 0; j < 64; ++j){ float x = XD[tid*64 + j]; s += x; s2 += x*x; }
    float mu  = s * (1.f/64.f);
    float var = s2 * (1.f/64.f) - mu*mu;
    float inv = rsqrtf(var + 1e-5f);
#pragma unroll
    for (int j = 0; j < 64; ++j){
      float hx = (XD[tid*64 + j] - mu) * inv * ln_g[j] + ln_b[j];
      RC[tid*64 + j] = (_Float16)hx;
    }
  }
  for (int i = tid; i < 64*80; i += 256){
    int k = i / 80, col = i - k*80;
    float w = (col < 32) ? sem_w[k*32 + col]
            : (col < 64) ? epi_w[k*32 + (col-32)]
            : (col == 64) ? rtr_w[k] : 0.f;
    WB[i] = (_Float16)w;
  }
  __syncthreads();

  // ---- matmul3: h(64x64) @ [sem|epi|rtr|pad](64x80) -> tables (4x5 tiles)
#pragma unroll
  for (int tt = 0; tt < 3; ++tt){
    int tile = wv + tt*8;
    if (tile < 20){
      int r0 = (tile / 5) * 16, c0 = (tile % 5) * 16;
      v8f c = {};
#pragma unroll
      for (int ks = 0; ks < 64; ks += 32){
        v16h a = load_a_frag(RC, 64, r0, ks, lane);
        v16h b = load_b_frag(WB, 80, ks, c0, lane);
        c = __builtin_amdgcn_wmma_f32_16x16x32_f16(false, a, false, b, (short)0, c, false, false);
      }
      int n = lane & 15, mh = (lane >> 4) << 3;
#pragma unroll
      for (int r = 0; r < 8; ++r){
        int m = r0 + r + mh, col = c0 + n;
        float vsum = c[r];
        if (col < 32){
          float o = vsum + sem_b[col];
          ws_ks[m*32 + col] = o; XD[m*32 + col] = o;             // Pks
        } else if (col < 64){
          int cc = col - 32;
          float o = vsum + epi_b[cc];
          ws_ke[m*32 + cc] = o; XD[2048 + m*32 + cc] = o;        // Pke
        } else if (col == 64){
          float z = vsum + rtr_b[0];
          GB[m] = 1.f / (1.f + __expf(-z));
        }
      }
    }
  }
  __syncthreads();

  if (tid < 64){
    float ss = 0.f, se = 0.f;
#pragma unroll
    for (int j = 0; j < 32; ++j){
      float a = XD[tid*32 + j];        ss += a*a;
      float b = XD[2048 + tid*32 + j]; se += b*b;
    }
    float g = GB[tid];
    ws_scal[tid*4 + 0] = g;
    ws_scal[tid*4 + 1] = 1.f - g;
    ws_scal[tid*4 + 2] = 1.f / (ss + 1e-6f);
    ws_scal[tid*4 + 3] = 1.f / (se + 1e-6f);
  }
}

// ---------------- Kernel 2: sequential delta-rule scan ----------------
// one block per batch; wave0 owns M_s, wave1 owns M_e; lane=row, regs=columns
__global__ __launch_bounds__(64) void scan_kernel(
    const int* __restrict__ seq,
    const float* __restrict__ ws_ks, const float* __restrict__ ws_ke,
    const float* __restrict__ ws_scal,
    const float* __restrict__ out_w, const float* __restrict__ out_b,
    float* __restrict__ outp)
{
  __shared__ __align__(16) float lks[64*32];
  __shared__ __align__(16) float lke[64*32];
  __shared__ __align__(16) float lsc[64*4];
  __shared__ float csce[64];

  const int tid  = threadIdx.x;
  const int lane = tid & 31;
  const int wv   = tid >> 5;
  const int b    = blockIdx.x;

  for (int i = tid; i < 2048; i += 64){ lks[i] = ws_ks[i]; lke[i] = ws_ke[i]; }
  for (int i = tid; i < 256;  i += 64){ lsc[i] = ws_scal[i]; }
  __syncthreads();

  const float* tab  = (wv == 0) ? lks : lke;
  const int*   seqb = seq + b * Ln;
  const float  invL = 1.f / (float)Ln;

  float M[32];
#pragma unroll
  for (int j = 0; j < 32; ++j) M[j] = 0.f;

  int tokv = 0;
  for (int t = 0; t < Ln - 1; ++t){
    if ((t & 31) == 0) tokv = seqb[t + lane];           // 32-token prefetch per wave
    int tok = __builtin_amdgcn_readlane(tokv, t & 31);  // uniform -> SGPR

    // broadcast k columns via uniform-address ds_load_b128 (all lanes identical)
    const float4* k4 = reinterpret_cast<const float4*>(tab + tok * 32);
    float kc[32];
#pragma unroll
    for (int q = 0; q < 8; ++q){
      float4 f = k4[q];
      kc[q*4+0] = f.x; kc[q*4+1] = f.y; kc[q*4+2] = f.z; kc[q*4+3] = f.w;
    }
    float  kl = tab[tok * 32 + lane];                    // k[row] per lane
    float4 sc = *reinterpret_cast<const float4*>(lsc + tok * 4);

    // vps[i] = sum_j M[i][j]*k[j]  (in-lane, 4 interleaved accumulators)
    float a0 = 0.f, a1 = 0.f, a2 = 0.f, a3 = 0.f;
#pragma unroll
    for (int j = 0; j < 32; j += 4){
      a0 = fmaf(M[j+0], kc[j+0], a0);
      a1 = fmaf(M[j+1], kc[j+1], a1);
      a2 = fmaf(M[j+2], kc[j+2], a2);
      a3 = fmaf(M[j+3], kc[j+3], a3);
    }
    float vps  = (a0 + a1) + (a2 + a3);
    float inv  = (wv == 0) ? sc.z : sc.w;                // 1/(sum k^2 + 1e-6), per token
    float dv   = fmaf(-vps, inv, kl);                    // k[i] - vps[i]*inv
    float rr   = (float)(t + 1) * invL;
    float coef = (wv == 0) ? sc.x : sc.y * rr;           // g   vs  (1-g)*r
    float tcf  = dv * coef;
#pragma unroll
    for (int j = 0; j < 32; ++j) M[j] = fmaf(tcf, kc[j], M[j]);   // rank-1 update
  }

  // readout: cs = M_s @ ks[last tok] (table already includes bias), ce likewise
  {
    int tokL = seqb[Ln - 1];
    const float4* k4 = reinterpret_cast<const float4*>(tab + tokL * 32);
    float kc[32];
#pragma unroll
    for (int q = 0; q < 8; ++q){
      float4 f = k4[q];
      kc[q*4+0] = f.x; kc[q*4+1] = f.y; kc[q*4+2] = f.z; kc[q*4+3] = f.w;
    }
    float a0 = 0.f, a1 = 0.f, a2 = 0.f, a3 = 0.f;
#pragma unroll
    for (int j = 0; j < 32; j += 4){
      a0 = fmaf(M[j+0], kc[j+0], a0);
      a1 = fmaf(M[j+1], kc[j+1], a1);
      a2 = fmaf(M[j+2], kc[j+2], a2);
      a3 = fmaf(M[j+3], kc[j+3], a3);
    }
    csce[wv * 32 + lane] = (a0 + a1) + (a2 + a3);
  }
  __syncthreads();

  // out[b][v] = concat(cs,ce) @ out_w + out_b   (64 threads = 64 outputs, coalesced)
  float acc = out_b[tid];
#pragma unroll 8
  for (int c2 = 0; c2 < 64; ++c2) acc = fmaf(csce[c2], out_w[c2 * 64 + tid], acc);
  outp[b * 64 + tid] = acc;
}

extern "C" void kernel_launch(void* const* d_in, const int* in_sizes, int n_in,
                              void* d_out, int out_size, void* d_ws, size_t ws_size,
                              hipStream_t stream)
{
  const int*   seq   = (const int*)  d_in[0];
  const float* embed = (const float*)d_in[1];
  const float* ff_w1 = (const float*)d_in[2];
  const float* ff_b1 = (const float*)d_in[3];
  const float* ff_w2 = (const float*)d_in[4];
  const float* ff_b2 = (const float*)d_in[5];
  const float* ln_g  = (const float*)d_in[6];
  const float* ln_b  = (const float*)d_in[7];
  const float* sem_w = (const float*)d_in[8];
  const float* sem_b = (const float*)d_in[9];
  const float* epi_w = (const float*)d_in[10];
  const float* epi_b = (const float*)d_in[11];
  const float* rtr_w = (const float*)d_in[12];
  const float* rtr_b = (const float*)d_in[13];
  const float* out_w = (const float*)d_in[14];
  const float* out_b = (const float*)d_in[15];

  float* ws_ks   = (float*)d_ws;          // 64*32
  float* ws_ke   = ws_ks + 64*32;         // 64*32
  float* ws_scal = ws_ke + 64*32;         // 64*4

  build_table_kernel<<<1, 256, 0, stream>>>(embed, ff_w1, ff_b1, ff_w2, ff_b2,
      ln_g, ln_b, sem_w, sem_b, epi_w, epi_b, rtr_w, rtr_b,
      ws_ks, ws_ke, ws_scal);
  scan_kernel<<<Bn, 64, 0, stream>>>(seq, ws_ks, ws_ke, ws_scal, out_w, out_b,
      (float*)d_out);
}
